// AttentionHead_46909632807549
// MI455X (gfx1250) — matrix operand order repaired and checked by reading, not compiled, not run on hardware
//
#include <hip/hip_runtime.h>
#include <math.h>

typedef __attribute__((ext_vector_type(16))) __bf16 v16bf;
typedef __attribute__((ext_vector_type(8)))  float  v8f;

#define EMB 768
#define HD  64
#define SEQ 2048
#define NB  4

__device__ __forceinline__ v8f wmma_bf16(v16bf a, v16bf b, v8f c) {
    // D = A(16x32 bf16) x B(32x16 bf16) + C(16x16 f32)
    return __builtin_amdgcn_wmma_f32_16x16x32_bf16(
        /*neg_a=*/false, a, /*neg_b=*/false, b,
        /*c_mod=*/(short)0, c, /*reuse_a=*/false, /*reuse_b=*/false);
}

// A-fragment K index for element e, half = lane>>4 (contiguous: k = 16*half + e)
__device__ __forceinline__ int kmapA(int e, int half) {
    return e + (half << 4);
}

// ---------------------------------------------------------------------------
// Kernel 1: convert Wq/Wk/Wv (fp32 [768][64]) -> bf16 transposed [3][64][768]
// so projection B-fragments are contiguous 32B per lane.
// ---------------------------------------------------------------------------
__global__ void __launch_bounds__(256) convert_w_kernel(
    const float* __restrict__ Wq, const float* __restrict__ Wk,
    const float* __restrict__ Wv, __bf16* __restrict__ WbfT) {
    int i = blockIdx.x * 256 + threadIdx.x;
    if (i >= 3 * HD * EMB) return;
    int w = i / (HD * EMB);
    int r = i - w * HD * EMB;
    int n = r / EMB;          // output column (0..63)
    int k = r - n * EMB;      // reduction index (0..767)
    const float* W = (w == 0) ? Wq : (w == 1) ? Wk : Wv;
    WbfT[i] = (__bf16)W[k * HD + n];
}

// ---------------------------------------------------------------------------
// Kernel 2: fused QKV projection.
// One wave per (16-row tile of B*S) x (q|k|v). 1536 waves total.
// C = H[16x768] @ W[768x64] + bias, via 24 K-steps of wmma bf16.
// Emits bf16: Q [BS][64], K [BS][64], V transposed [B][64][S].
// ---------------------------------------------------------------------------
__global__ void __launch_bounds__(256) qkv_proj_kernel(
    const float* __restrict__ H,
    const float* __restrict__ bq, const float* __restrict__ bk,
    const float* __restrict__ bv,
    const __bf16* __restrict__ WbfT,
    __bf16* __restrict__ Qbf, __bf16* __restrict__ Kbf,
    __bf16* __restrict__ Vt) {
    const int lane = threadIdx.x & 31;
    const int wid  = threadIdx.x >> 5;
    const int half = lane >> 4;
    const int lrow = lane & 15;

    int g    = blockIdx.x * 8 + wid;   // 0..1535
    int tile = g / 3;                  // 0..511 : 16-row tile of flattened B*S
    int wsel = g - tile * 3;           // 0=q 1=k 2=v

    const int arow = tile * 16 + lrow; // A-fragment row (M = lane&15)

    v8f acc[4] = {};                   // 16 x 64 fp32 accumulator

    for (int ks = 0; ks < EMB / 32; ++ks) {
        const int kb = ks * 32;
        // A: hidden_state fp32 -> bf16 on the fly (each element read once)
        v16bf a;
#pragma unroll
        for (int e = 0; e < 16; ++e)
            a[e] = (__bf16)H[arow * EMB + kb + kmapA(e, half)];
#pragma unroll
        for (int nt = 0; nt < 4; ++nt) {
            // B: contiguous 32B per lane from transposed weights
            const v16bf b = *(const v16bf*)(
                WbfT + (wsel * HD + nt * 16 + lrow) * EMB + kb + half * 16);
            acc[nt] = wmma_bf16(a, b, acc[nt]);
        }
    }

    const float* bias = (wsel == 0) ? bq : (wsel == 1) ? bk : bv;
#pragma unroll
    for (int nt = 0; nt < 4; ++nt) {
#pragma unroll
        for (int r = 0; r < 8; ++r) {
            const int m    = r + half * 8;           // C/D row
            const int n    = nt * 16 + lrow;         // C/D col
            const int orow = tile * 16 + m;          // flattened B*S row
            const __bf16 v = (__bf16)(acc[nt][r] + bias[n]);
            if (wsel == 0)      Qbf[orow * HD + n] = v;
            else if (wsel == 1) Kbf[orow * HD + n] = v;
            else {
                const int bb = orow >> 11;           // / SEQ
                const int ss = orow & (SEQ - 1);
                Vt[(bb * HD + n) * SEQ + ss] = v;    // transposed for AV B-frags
            }
        }
    }
}

// ---------------------------------------------------------------------------
// Kernel 3: flash attention. One wave per (batch, 16-query tile); 512 waves.
// Per 32-key block: 4 score WMMAs, online softmax (shfl_xor row reductions),
// P bounced through 1KB private LDS into A-fragment layout, 4 AV WMMAs.
// ---------------------------------------------------------------------------
__global__ void __launch_bounds__(256) attn_kernel(
    const __bf16* __restrict__ Qbf, const __bf16* __restrict__ Kbf,
    const __bf16* __restrict__ Vt, float* __restrict__ out) {
    __shared__ __bf16 sh[8 * 16 * 32];           // 1KB per wave
    const int lane = threadIdx.x & 31;
    const int wid  = threadIdx.x >> 5;
    const int half = lane >> 4;
    const int lrow = lane & 15;

    const int qt      = blockIdx.x * 8 + wid;    // 0..511
    const int b       = qt >> 7;                 // 128 q-tiles per batch
    const int qtile   = qt & 127;
    const int rowbase = b * SEQ + qtile * 16;    // flattened B*S
    __bf16* psh = sh + wid * 512;

    const __bf16* Kb = Kbf + (size_t)b * SEQ * HD;
    const __bf16* Vb = Vt  + (size_t)b * HD * SEQ;

    // Q A-fragments, d = 0..31 and 32..63
    v16bf aq0, aq1;
#pragma unroll
    for (int e = 0; e < 16; ++e) {
        const int r64 = (rowbase + lrow) * HD;
        aq0[e] = Qbf[r64 + kmapA(e, half)];
        aq1[e] = Qbf[r64 + 32 + kmapA(e, half)];
    }

    v8f acc[4] = {};
    float mrun[8], lrun[8];
#pragma unroll
    for (int r = 0; r < 8; ++r) { mrun[r] = -1e30f; lrun[r] = 0.0f; }

    for (int jb = 0; jb < SEQ; jb += 32) {
        // ---- scores: two 16-key tiles, 16x16 each (Q @ K^T) ----
        v8f s0 = {}, s1 = {};
        {
            const v16bf bk0a = *(const v16bf*)(Kb + (jb + lrow) * HD + half * 16);
            const v16bf bk0b = *(const v16bf*)(Kb + (jb + lrow) * HD + 32 + half * 16);
            s0 = wmma_bf16(aq0, bk0a, s0);
            s0 = wmma_bf16(aq1, bk0b, s0);
            const v16bf bk1a = *(const v16bf*)(Kb + (jb + 16 + lrow) * HD + half * 16);
            const v16bf bk1b = *(const v16bf*)(Kb + (jb + 16 + lrow) * HD + 32 + half * 16);
            s1 = wmma_bf16(aq0, bk1a, s1);
            s1 = wmma_bf16(aq1, bk1b, s1);
        }

        // ---- online softmax; one row lives in 16 lanes of one half-wave ----
        const float scale = 0.125f;                  // 1/sqrt(64)
#pragma unroll
        for (int r = 0; r < 8; ++r) {
            const float x0 = s0[r] * scale;
            const float x1 = s1[r] * scale;
            float mt = fmaxf(x0, x1);
#pragma unroll
            for (int off = 1; off < 16; off <<= 1)
                mt = fmaxf(mt, __shfl_xor(mt, off, 32));
            const float mnew = fmaxf(mrun[r], mt);
            const float f    = __expf(mrun[r] - mnew);
            mrun[r] = mnew;
            const float p0 = __expf(x0 - mnew);
            const float p1 = __expf(x1 - mnew);
            float rs = p0 + p1;
#pragma unroll
            for (int off = 1; off < 16; off <<= 1)
                rs += __shfl_xor(rs, off, 32);
            lrun[r] = lrun[r] * f + rs;
            acc[0][r] *= f; acc[1][r] *= f; acc[2][r] *= f; acc[3][r] *= f;
            // P (C-layout) -> LDS as row-major 16x32 bf16
            psh[(r + half * 8) * 32 + lrow]      = (__bf16)p0;
            psh[(r + half * 8) * 32 + 16 + lrow] = (__bf16)p1;
        }

        // ---- P back as A-fragment (same-wave LDS is in-order, no barrier) --
        v16bf ap;
#pragma unroll
        for (int e = 0; e < 16; ++e)
            ap[e] = psh[lrow * 32 + kmapA(e, half)];

        // ---- AV: out(16x64) += P(16x32) @ V(32x64), V^T gives 32B loads ----
#pragma unroll
        for (int nt = 0; nt < 4; ++nt) {
            const v16bf bv = *(const v16bf*)(
                Vb + (nt * 16 + lrow) * SEQ + jb + half * 16);
            acc[nt] = wmma_bf16(ap, bv, acc[nt]);
        }
    }

    // ---- epilogue: normalize by running denominator, store fp32 ----
#pragma unroll
    for (int r = 0; r < 8; ++r) {
        const float inv  = 1.0f / lrun[r];
        const int   orow = rowbase + r + half * 8;
#pragma unroll
        for (int nt = 0; nt < 4; ++nt)
            out[orow * HD + nt * 16 + lrow] = acc[nt][r] * inv;
    }
}

// ---------------------------------------------------------------------------
extern "C" void kernel_launch(void* const* d_in, const int* in_sizes, int n_in,
                              void* d_out, int out_size, void* d_ws, size_t ws_size,
                              hipStream_t stream) {
    const float* H  = (const float*)d_in[0];
    const float* Wq = (const float*)d_in[1];
    const float* bq = (const float*)d_in[2];
    const float* Wk = (const float*)d_in[3];
    const float* bk = (const float*)d_in[4];
    const float* Wv = (const float*)d_in[5];
    const float* bv = (const float*)d_in[6];
    float* out = (float*)d_out;

    char* ws = (char*)d_ws;
    const size_t szW = (size_t)3 * HD * EMB * 2;        // 294912 B (32B aligned)
    const size_t szQ = (size_t)NB * SEQ * HD * 2;       // 1 MiB each
    __bf16* WbfT = (__bf16*)(ws);
    __bf16* Qbf  = (__bf16*)(ws + szW);
    __bf16* Kbf  = (__bf16*)(ws + szW + szQ);
    __bf16* Vt   = (__bf16*)(ws + szW + 2 * szQ);

    convert_w_kernel<<<(3 * HD * EMB + 255) / 256, 256, 0, stream>>>(Wq, Wk, Wv, WbfT);
    // 512 row-tiles x {q,k,v} = 1536 waves, 8 waves/block
    qkv_proj_kernel<<<192, 256, 0, stream>>>(H, bq, bk, bv, WbfT, Qbf, Kbf, Vt);
    // 4 batches x 128 query tiles = 512 waves, 8 waves/block
    attn_kernel<<<64, 256, 0, stream>>>(Qbf, Kbf, Vt, out);
}
